// PaliGemmaWithExpert_3427383903005
// MI455X (gfx1250) — matrix operand order repaired and checked by reading, not compiled, not run on hardware
//
#include <hip/hip_runtime.h>
#include <hip/hip_bf16.h>
#include <cstdint>

// ---------------- types ----------------
typedef __attribute__((ext_vector_type(16))) __bf16 v16bf;
typedef __attribute__((ext_vector_type(8)))  __bf16 bf16x8;
typedef __attribute__((ext_vector_type(8)))  float  v8f;

// ---------------- problem constants ----------------
#define Bsz 4
#define SPp 768
#define SSs 256
#define Tt  1024     // SPp + SSs
#define WPw 2048
#define WSw 1024
#define NH  8
#define HDh 256
#define ATT 2048     // NH * HDh

// ---------------- CDNA5 async global->LDS copy (ASYNCcnt path) ----------------
__device__ __forceinline__ void async_ld_b128(unsigned lds_byte_off, const void* g) {
  asm volatile("global_load_async_to_lds_b128 %0, %1, off"
               :: "v"(lds_byte_off), "v"(g) : "memory");
}
__device__ __forceinline__ void wait_async_le0() {
  asm volatile("s_wait_asynccnt 0" ::: "memory");
}
__device__ __forceinline__ void wait_async_le4() {
  asm volatile("s_wait_asynccnt 4" ::: "memory");   // 4 in-flight (next tile) allowed
}
__device__ __forceinline__ unsigned lds_off(const void* p) {
  return (unsigned)(unsigned long long)p;   // low 32 bits of generic LDS address
}

// ---------------- WMMA helpers (CDNA5 16x16x32 bf16, f32 acc) ----------------
__device__ __forceinline__ v8f wmma_bf16(v16bf a, v16bf b, v8f c) {
  return __builtin_amdgcn_wmma_f32_16x16x32_bf16(false, a, false, b, (short)0, c,
                                                 false, false);
}

// A fragment: 16x32 bf16, row-major source (element(m,k) at base[m*stride+k]).
// ISA layout: lanes 0-15 row M=lane, K {0..7, 16..23}; lanes 16-31 K {8..15, 24..31}.
__device__ __forceinline__ v16bf load_a_frag(const __bf16* base, int stride, int lane) {
  int row = lane & 15;
  int kb  = (lane >> 4) * 8;
  const __bf16* p = base + (size_t)row * stride + kb;
  bf16x8 lo = *(const bf16x8*)(p);
  bf16x8 hi = *(const bf16x8*)(p + 16);
  v16bf a;
  #pragma unroll
  for (int i = 0; i < 8; ++i) { a[i] = lo[i]; a[8 + i] = hi[i]; }
  return a;
}

// B fragment: 32x16 bf16, K-contiguous source (element(k,n) at base[n*stride+k]).
// ISA layout: lanes 0-15 K {0..7, 16..23}, lanes 16-31 K {8..15, 24..31}, col N = lane%16.
__device__ __forceinline__ v16bf load_b_frag_t(const __bf16* base, int stride, int lane) {
  int col = lane & 15;
  int kb  = (lane >> 4) * 8;
  const __bf16* p = base + (size_t)col * stride + kb;
  bf16x8 lo = *(const bf16x8*)(p);
  bf16x8 hi = *(const bf16x8*)(p + 16);
  v16bf b;
  #pragma unroll
  for (int i = 0; i < 8; ++i) { b[i] = lo[i]; b[8 + i] = hi[i]; }
  return b;
}

// ---------------- fused cast + transpose: X (K x N, f32) -> Yt (N x K, bf16) -----
__global__ __launch_bounds__(256)
void cast_transpose_bf16_kernel(const float* __restrict__ X, __bf16* __restrict__ Yt,
                                int K, int N) {
  __shared__ float tile[32][33];
  const int kt = blockIdx.y * 32;
  const int nt = blockIdx.x * 32;
  const int tx = threadIdx.x & 31;
  const int ty = threadIdx.x >> 5;
  #pragma unroll
  for (int r = ty; r < 32; r += 8)
    tile[r][tx] = X[(size_t)(kt + r) * N + nt + tx];
  __syncthreads();
  #pragma unroll
  for (int r = ty; r < 32; r += 8)
    Yt[(size_t)(nt + r) * K + kt + tx] = (__bf16)tile[tx][r];
}

// ---------------- RMSNorm (fp32 in -> bf16 out) ----------------
__global__ __launch_bounds__(256)
void rmsnorm_bf16_kernel(const float* __restrict__ X, const float* __restrict__ W,
                         __bf16* __restrict__ Hout, int width) {
  const int row  = blockIdx.x;
  const int tid  = threadIdx.x;
  const int lane = tid & 31;
  const int wv   = tid >> 5;
  const float* xr = X + (size_t)row * width;
  __shared__ float red[8];
  __shared__ float rsv;
  float s = 0.f;
  for (int i = tid; i < width; i += 256) { float v = xr[i]; s += v * v; }
  #pragma unroll
  for (int off = 16; off; off >>= 1) s += __shfl_xor(s, off, 32);
  if (lane == 0) red[wv] = s;
  __syncthreads();
  if (tid == 0) {
    float t = 0.f;
    for (int i = 0; i < 8; ++i) t += red[i];
    rsv = rsqrtf(t / (float)width + 1e-6f);
  }
  __syncthreads();
  const float rs = rsv;
  __bf16* hr = Hout + (size_t)row * width;
  for (int i = tid; i < width; i += 256)
    hr[i] = (__bf16)(xr[i] * rs * (1.f + W[i]));
}

// ---------------- bf16 WMMA GEMM, 128x128 tile, double-buffered async staging ----
// C[M,N](f32) = mapA(A)[M,K](bf16) @ Bt[N,K]^T(bf16) (+ Res)
// A row remap: ar = (r/map_div)*map_mul + r%map_div + map_add (identity: map_div=1<<30).
// Grid: (N/128, M/128). Block: 256 threads (8 waves); wave = 32 rows x 64 cols.
__global__ __launch_bounds__(256)
void gemm_bf16_kernel(const __bf16* __restrict__ A, int lda,
                      const __bf16* __restrict__ Bt, int ldbt,
                      float* __restrict__ C, const float* __restrict__ Res, int ldc,
                      int K, int map_div, int map_mul, int map_add) {
  const int tid  = threadIdx.x;
  const int lane = tid & 31;
  const int wv   = tid >> 5;
  const int rowBlk = blockIdx.y * 128;
  const int colBlk = blockIdx.x * 128;

  __shared__ __bf16 sA[2][128 * 32];    // [m][k]
  __shared__ __bf16 sBt[2][128 * 32];   // [n][k] (K-contiguous; no transpose needed)

  const int wr = (wv >> 1) * 32;    // wave row base in tile
  const int wc = (wv & 1) * 64;     // wave col base in tile

  v8f zero = {};
  v8f acc[2][4];
  #pragma unroll
  for (int rt = 0; rt < 2; ++rt)
    #pragma unroll
    for (int j = 0; j < 4; ++j) acc[rt][j] = zero;

  auto stage = [&](int buf, int kt) {
    #pragma unroll
    for (int i = 0; i < 2; ++i) {        // A tile: 128x32 = 512 x 16B chunks
      int c    = tid + i * 256;
      int row  = c >> 2;
      int part = (c & 3) * 8;
      int gr   = rowBlk + row;
      int ar   = (gr / map_div) * map_mul + (gr % map_div) + map_add;
      async_ld_b128(lds_off(&sA[buf][row * 32 + part]),
                    &A[(size_t)ar * lda + kt + part]);
    }
    #pragma unroll
    for (int i = 0; i < 2; ++i) {        // Bt tile: 128x32 = 512 x 16B chunks
      int c    = tid + i * 256;
      int n    = c >> 2;
      int part = (c & 3) * 8;
      async_ld_b128(lds_off(&sBt[buf][n * 32 + part]),
                    &Bt[(size_t)(colBlk + n) * ldbt + kt + part]);
    }
  };

  const int nk = K / 32;
  stage(0, 0);
  for (int it = 0; it < nk; ++it) {
    const int buf = it & 1;
    if (it + 1 < nk) {
      stage(buf ^ 1, (it + 1) * 32);     // prefetch next tile into other buffer
      wait_async_le4();                  // current tile's 4 chunks complete
    } else {
      wait_async_le0();
    }
    __syncthreads();
    v16bf a0 = load_a_frag(&sA[buf][(wr + 0) * 32], 32, lane);
    v16bf a1 = load_a_frag(&sA[buf][(wr + 16) * 32], 32, lane);
    #pragma unroll
    for (int j = 0; j < 4; ++j) {
      v16bf b = load_b_frag_t(&sBt[buf][(wc + j * 16) * 32], 32, lane);
      acc[0][j] = wmma_bf16(a0, b, acc[0][j]);
      acc[1][j] = wmma_bf16(a1, b, acc[1][j]);
    }
    __syncthreads();
  }

  const int col = lane & 15;
  const int rb  = (lane >> 4) * 8;
  #pragma unroll
  for (int rt = 0; rt < 2; ++rt) {
    #pragma unroll
    for (int j = 0; j < 4; ++j) {
      #pragma unroll
      for (int r = 0; r < 8; ++r) {
        int gr = rowBlk + wr + rt * 16 + rb + r;
        size_t o = (size_t)gr * ldc + colBlk + wc + j * 16 + col;
        float v = acc[rt][j][r];
        if (Res) v += Res[o];
        C[o] = v;
      }
    }
  }
}

// ---------------- RoPE + gather (also emits V^T for K-contiguous PV loads) ------
__global__ __launch_bounds__(256)
void rope_gather_kernel(const float* __restrict__ qpp, const float* __restrict__ qps,
                        const float* __restrict__ kpp, const float* __restrict__ kps,
                        const float* __restrict__ vpp, const float* __restrict__ vps,
                        const int* __restrict__ pos_ids,
                        __bf16* __restrict__ Q, __bf16* __restrict__ Kb,
                        __bf16* __restrict__ Vt) {
  const int bid = blockIdx.x;
  const int b = bid / Tt;
  const int t = bid % Tt;
  const int tid = threadIdx.x;
  const bool pre = (t < SPp);
  const float* qrow = pre ? qpp + (size_t)(b * SPp + t) * ATT
                          : qps + (size_t)(b * SSs + (t - SPp)) * ATT;
  const float* krow = pre ? kpp + (size_t)(b * SPp + t) * HDh
                          : kps + (size_t)(b * SSs + (t - SPp)) * HDh;
  const float* vrow = pre ? vpp + (size_t)(b * SPp + t) * HDh
                          : vps + (size_t)(b * SSs + (t - SPp)) * HDh;
  const float p = (float)pos_ids[b * Tt + t];

  for (int idx = tid; idx < ATT; idx += 256) {
    int h = idx >> 8, d = idx & 255;
    int j = d & 127;
    float freq = __powf(10000.f, -(float)(2 * j) * (1.f / 256.f));
    float ang = p * freq;
    float c = __cosf(ang), sn = __sinf(ang);
    float x  = qrow[idx];
    float xr = (d < 128) ? -qrow[idx + 128] : qrow[idx - 128];
    Q[(((size_t)(b * NH + h)) * Tt + t) * HDh + d] = (__bf16)(x * c + xr * sn);
  }
  for (int d = tid; d < HDh; d += 256) {
    int j = d & 127;
    float freq = __powf(10000.f, -(float)(2 * j) * (1.f / 256.f));
    float ang = p * freq;
    float c = __cosf(ang), sn = __sinf(ang);
    float x  = krow[d];
    float xr = (d < 128) ? -krow[d + 128] : krow[d - 128];
    Kb[((size_t)(b * Tt + t)) * HDh + d] = (__bf16)(x * c + xr * sn);
    Vt[((size_t)b * HDh + d) * Tt + t]  = (__bf16)vrow[d];   // transposed V
  }
}

// ---------------- attention: one (b,h,32-query block) per workgroup ----------------
// Dynamic LDS: sS 32x1024 f32 | sP 32x1024 bf16 | sQ 32x256 bf16 = 212992 B.
__global__ __launch_bounds__(256)
void attention_kernel(const __bf16* __restrict__ Q, const __bf16* __restrict__ Kb,
                      const __bf16* __restrict__ Vt, const float* __restrict__ mask,
                      __bf16* __restrict__ Out) {
  const int tid  = threadIdx.x;
  const int lane = tid & 31;
  const int wv   = tid >> 5;
  const int blk  = blockIdx.x;
  const int qb = blk & 31;          // 32 query blocks of 32
  const int h  = (blk >> 5) & 7;
  const int b  = blk >> 8;
  const int q0 = qb * 32;

  extern __shared__ char smem[];
  float*  sS = (float*)smem;                               // 32 x 1024
  __bf16* sP = (__bf16*)(smem + 32 * 1024 * 4);            // 32 x 1024
  __bf16* sQ = (__bf16*)(smem + 32 * 1024 * 6);            // 32 x 256

  // stage the query block via async copy (32x256 bf16 = 1024 x 16B chunks)
  const __bf16* qsrc = Q + ((size_t)(b * NH + h) * Tt + q0) * HDh;
  #pragma unroll
  for (int i = 0; i < 4; ++i) {
    int c = tid + i * 256;
    async_ld_b128(lds_off(&sQ[c * 8]), qsrc + (size_t)c * 8);
  }
  wait_async_le0();
  __syncthreads();

  // ---- S = Q @ K^T (K rows are K-contiguous per key -> vector B frags) ----
  const __bf16* Kbase = Kb + (size_t)b * Tt * HDh;
  for (int ct = wv; ct < 64; ct += 8) {
    int n0 = ct * 16;
    #pragma unroll
    for (int rt = 0; rt < 2; ++rt) {
      v8f acc = {};
      #pragma unroll
      for (int kk = 0; kk < HDh; kk += 32) {
        v16bf a  = load_a_frag(sQ + rt * 16 * HDh + kk, HDh, lane);
        v16bf bb = load_b_frag_t(Kbase + (size_t)n0 * HDh + kk, HDh, lane);
        acc = wmma_bf16(a, bb, acc);
      }
      int col = lane & 15;
      int rb  = (lane >> 4) * 8;
      #pragma unroll
      for (int r = 0; r < 8; ++r)
        sS[(rt * 16 + rb + r) * 1024 + n0 + col] = acc[r];
    }
  }
  __syncthreads();

  // ---- fp32 softmax (4 rows per wave) ----
  const float scale = 0.0625f;   // HD^-0.5
  const float* mbase = mask + ((size_t)b * Tt + q0) * Tt;
  for (int r = wv * 4; r < wv * 4 + 4; ++r) {
    float* srow = sS + r * 1024;
    const float* mr = mbase + (size_t)r * Tt;
    float mx = -1e30f;
    for (int c = lane; c < 1024; c += 32) {
      float v = srow[c] * scale + mr[c];
      srow[c] = v;
      mx = fmaxf(mx, v);
    }
    #pragma unroll
    for (int off = 16; off; off >>= 1) mx = fmaxf(mx, __shfl_xor(mx, off, 32));
    float sum = 0.f;
    for (int c = lane; c < 1024; c += 32) {
      float e = __expf(srow[c] - mx);
      srow[c] = e;
      sum += e;
    }
    #pragma unroll
    for (int off = 16; off; off >>= 1) sum += __shfl_xor(sum, off, 32);
    float inv = 1.f / sum;
    __bf16* prow = sP + r * 1024;
    for (int c = lane; c < 1024; c += 32) prow[c] = (__bf16)(srow[c] * inv);
  }
  __syncthreads();

  // ---- O = P @ V: V^T gives K-contiguous global vector B frags, no staging ----
  const __bf16* Vtb = Vt + (size_t)b * HDh * Tt;
  __bf16* obase = Out + ((size_t)b * Tt + q0) * ATT + h * HDh;
  const int c0 = wv * 32;
  v8f zero = {};
  v8f acc0[2] = {zero, zero};
  v8f acc1[2] = {zero, zero};
  for (int kk = 0; kk < Tt; kk += 32) {
    #pragma unroll
    for (int rt = 0; rt < 2; ++rt) {
      v16bf a  = load_a_frag(sP + rt * 16 * 1024 + kk, 1024, lane);
      v16bf b0 = load_b_frag_t(Vtb + (size_t)(c0 + 0) * Tt + kk, Tt, lane);
      v16bf b1 = load_b_frag_t(Vtb + (size_t)(c0 + 16) * Tt + kk, Tt, lane);
      acc0[rt] = wmma_bf16(a, b0, acc0[rt]);
      acc1[rt] = wmma_bf16(a, b1, acc1[rt]);
    }
  }
  {
    int col = lane & 15;
    int rb  = (lane >> 4) * 8;
    #pragma unroll
    for (int rt = 0; rt < 2; ++rt) {
      #pragma unroll
      for (int r = 0; r < 8; ++r) {
        int t = rt * 16 + rb + r;
        obase[(size_t)t * ATT + c0 + col]      = (__bf16)acc0[rt][r];
        obase[(size_t)t * ATT + c0 + 16 + col] = (__bf16)acc1[rt][r];
      }
    }
  }
}

// ---------------- host ----------------
static inline void* bump(char*& p, size_t bytes) {
  void* r = (void*)p;
  p += (bytes + 255) & ~(size_t)255;
  return r;
}

extern "C" void kernel_launch(void* const* d_in, const int* in_sizes, int n_in,
                              void* d_out, int out_size, void* d_ws, size_t ws_size,
                              hipStream_t stream) {
  (void)in_sizes; (void)n_in; (void)out_size; (void)ws_size;
  const float* prefix_x = (const float*)d_in[0];
  const float* suffix_x = (const float*)d_in[1];
  const float* mask     = (const float*)d_in[2];
  const int*   pos      = (const int*)d_in[3];
  const float* ln_p = (const float*)d_in[4];
  const float* q_p  = (const float*)d_in[5];
  const float* k_p  = (const float*)d_in[6];
  const float* v_p  = (const float*)d_in[7];
  const float* o_p  = (const float*)d_in[8];
  const float* ln_s = (const float*)d_in[9];
  const float* q_s  = (const float*)d_in[10];
  const float* k_s  = (const float*)d_in[11];
  const float* v_s  = (const float*)d_in[12];
  const float* o_s  = (const float*)d_in[13];
  float* out = (float*)d_out;

  char* wp = (char*)d_ws;
  const size_t MP = (size_t)Bsz * SPp;   // 3072 prefix rows
  const size_t MS = (size_t)Bsz * SSs;   // 1024 suffix rows

  __bf16* h_p  = (__bf16*)bump(wp, MP * WPw * 2);
  __bf16* h_s  = (__bf16*)bump(wp, MS * WSw * 2);
  __bf16* wqp  = (__bf16*)bump(wp, (size_t)WPw * ATT * 2);   // transposed: ATT x WPw
  __bf16* wkp  = (__bf16*)bump(wp, (size_t)WPw * HDh * 2);   // HDh x WPw
  __bf16* wvp  = (__bf16*)bump(wp, (size_t)WPw * HDh * 2);
  __bf16* wqs  = (__bf16*)bump(wp, (size_t)WSw * ATT * 2);   // ATT x WSw
  __bf16* wks  = (__bf16*)bump(wp, (size_t)WSw * HDh * 2);
  __bf16* wvs  = (__bf16*)bump(wp, (size_t)WSw * HDh * 2);
  __bf16* wop  = (__bf16*)bump(wp, (size_t)ATT * WPw * 2);   // WPw x ATT
  __bf16* wos  = (__bf16*)bump(wp, (size_t)ATT * WSw * 2);   // WSw x ATT
  float* qpre_p = (float*)bump(wp, MP * ATT * 4);
  float* qpre_s = (float*)bump(wp, MS * ATT * 4);
  float* kpre_p = (float*)bump(wp, MP * HDh * 4);
  float* kpre_s = (float*)bump(wp, MS * HDh * 4);
  float* vpre_p = (float*)bump(wp, MP * HDh * 4);
  float* vpre_s = (float*)bump(wp, MS * HDh * 4);
  __bf16* qbuf  = (__bf16*)bump(wp, (size_t)Bsz * NH * Tt * HDh * 2);
  __bf16* kbuf  = (__bf16*)bump(wp, (size_t)Bsz * Tt * HDh * 2);
  __bf16* vtbuf = (__bf16*)bump(wp, (size_t)Bsz * HDh * Tt * 2);
  __bf16* attn  = (__bf16*)bump(wp, (size_t)Bsz * Tt * ATT * 2);

  const int IDD = 1 << 30;   // identity row-map

  // --- weight cast + transpose (once; GEMMs then need no LDS transpose) ---
  auto castT = [&](const float* src, __bf16* dstT, int K, int N) {
    cast_transpose_bf16_kernel<<<dim3(N / 32, K / 32), 256, 0, stream>>>(src, dstT, K, N);
  };
  castT(q_p, wqp, WPw, ATT);
  castT(k_p, wkp, WPw, HDh);
  castT(v_p, wvp, WPw, HDh);
  castT(q_s, wqs, WSw, ATT);
  castT(k_s, wks, WSw, HDh);
  castT(v_s, wvs, WSw, HDh);
  castT(o_p, wop, ATT, WPw);
  castT(o_s, wos, ATT, WSw);

  rmsnorm_bf16_kernel<<<(unsigned)MP, 256, 0, stream>>>(prefix_x, ln_p, h_p, WPw);
  rmsnorm_bf16_kernel<<<(unsigned)MS, 256, 0, stream>>>(suffix_x, ln_s, h_s, WSw);

  // --- QKV GEMMs (128x128 tiles, Bt stride = K) ---
  gemm_bf16_kernel<<<dim3(ATT / 128, MP / 128), 256, 0, stream>>>(
      h_p, WPw, wqp, WPw, qpre_p, nullptr, ATT, WPw, IDD, 0, 0);
  gemm_bf16_kernel<<<dim3(HDh / 128, MP / 128), 256, 0, stream>>>(
      h_p, WPw, wkp, WPw, kpre_p, nullptr, HDh, WPw, IDD, 0, 0);
  gemm_bf16_kernel<<<dim3(HDh / 128, MP / 128), 256, 0, stream>>>(
      h_p, WPw, wvp, WPw, vpre_p, nullptr, HDh, WPw, IDD, 0, 0);
  gemm_bf16_kernel<<<dim3(ATT / 128, MS / 128), 256, 0, stream>>>(
      h_s, WSw, wqs, WSw, qpre_s, nullptr, ATT, WSw, IDD, 0, 0);
  gemm_bf16_kernel<<<dim3(HDh / 128, MS / 128), 256, 0, stream>>>(
      h_s, WSw, wks, WSw, kpre_s, nullptr, HDh, WSw, IDD, 0, 0);
  gemm_bf16_kernel<<<dim3(HDh / 128, MS / 128), 256, 0, stream>>>(
      h_s, WSw, wvs, WSw, vpre_s, nullptr, HDh, WSw, IDD, 0, 0);

  rope_gather_kernel<<<Bsz * Tt, 256, 0, stream>>>(
      qpre_p, qpre_s, kpre_p, kpre_s, vpre_p, vpre_s, pos, qbuf, kbuf, vtbuf);

  const size_t attn_lds = 32 * 1024 * 4 + 32 * 1024 * 2 + 32 * HDh * 2;  // 212992 B
  attention_kernel<<<Bsz * NH * (Tt / 32), 256, attn_lds, stream>>>(
      qbuf, kbuf, vtbuf, mask, attn);

  // --- output projections + residual (rows gathered out of the (B,T,ATT) buffer) ---
  gemm_bf16_kernel<<<dim3(WPw / 128, MP / 128), 256, 0, stream>>>(
      attn, ATT, wop, ATT, out, prefix_x, WPw, ATT, SPp, Tt, 0);
  gemm_bf16_kernel<<<dim3(WSw / 128, MS / 128), 256, 0, stream>>>(
      attn, ATT, wos, ATT, out + MP * (size_t)WPw, suffix_x, WSw, ATT, SSs, Tt, SPp);
}
